// RGCN_13675175870760
// MI455X (gfx1250) — matrix operand order repaired and checked by reading, not compiled, not run on hardware
//
#include <hip/hip_runtime.h>
#include <hip/hip_bf16.h>

typedef __attribute__((ext_vector_type(16))) _Float16 v16h;
typedef __attribute__((ext_vector_type(8)))  _Float16 v8h;
typedef __attribute__((ext_vector_type(8)))  float    v8f;

#define N_REL 8
#define HID 16

// ---------------------------------------------------------------------------
// Generic float zero
// ---------------------------------------------------------------------------
__global__ __launch_bounds__(256) void zero_f32(float* __restrict__ p, long long n) {
    long long i = (long long)blockIdx.x * 256 + threadIdx.x;
    if (i < n) p[i] = 0.0f;
}

// ---------------------------------------------------------------------------
// Pack B operand for GEMM1 into WMMA fragment layout.
// Bcat1[K,N] : K=0..127, N=0..143  (N<128 -> W1[r=N/16][K][o=N%16], else root1)
// Layout: Bp[((kt*9+nt)*32+lane)*16 + e], K = kt*32 + (lane>=16?8:0) + (e&7) + (e>=8?16:0),
//         N = nt*16 + (lane&15)
// ---------------------------------------------------------------------------
__global__ __launch_bounds__(256) void pack_b1(const float* __restrict__ W1,
                                               const float* __restrict__ root1,
                                               _Float16* __restrict__ Bp) {
    int idx = blockIdx.x * 256 + threadIdx.x;
    if (idx >= 4 * 9 * 32 * 16) return;
    int e    = idx & 15;
    int lane = (idx >> 4) & 31;
    int nt   = (idx >> 9) % 9;
    int kt   = idx / (9 * 512);
    int K = kt * 32 + ((lane >> 4) << 3) + (e & 7) + ((e >> 3) << 4);   // < 128
    int N = nt * 16 + (lane & 15);              // < 144
    float v;
    if (N < 128) {
        int r = N >> 4, o = N & 15;
        v = W1[((size_t)r * 128 + K) * 16 + o];
    } else {
        v = root1[(size_t)K * 16 + (N - 128)];
    }
    Bp[idx] = (_Float16)v;
}

// ---------------------------------------------------------------------------
// Pack B operand for GEMM2: Bcat2[K,N] K=0..159 (>=144 zero-pad), N=0..63
// K<128 -> W2[r=K/16][d=K%16][N]; 128<=K<144 -> root2[K-128][N]; else 0
// ---------------------------------------------------------------------------
__global__ __launch_bounds__(256) void pack_b2(const float* __restrict__ W2,
                                               const float* __restrict__ root2,
                                               _Float16* __restrict__ Bp) {
    int idx = blockIdx.x * 256 + threadIdx.x;
    if (idx >= 5 * 4 * 32 * 16) return;
    int e    = idx & 15;
    int lane = (idx >> 4) & 31;
    int nt   = (idx >> 9) % 4;
    int kt   = idx / (4 * 512);
    int K = kt * 32 + ((lane >> 4) << 3) + (e & 7) + ((e >> 3) << 4);   // < 160
    int N = nt * 16 + (lane & 15);              // < 64
    float v = 0.0f;
    if (K < 128) {
        int r = K >> 4, d = K & 15;
        v = W2[((size_t)r * 16 + d) * 64 + N];
    } else if (K < 144) {
        v = root2[(size_t)(K - 128) * 64 + N];
    }
    Bp[idx] = (_Float16)v;
}

// ---------------------------------------------------------------------------
// GEMM1: XW[100000,144] = X[100000,128] * Bcat1[128,144]   (f16 in, f32 acc)
// One wave32 per 16-row tile; 9 N-tiles x 4 K-steps of v_wmma_f32_16x16x32_f16
// ---------------------------------------------------------------------------
__global__ __launch_bounds__(256) void gemm1(const float* __restrict__ X,
                                             const _Float16* __restrict__ Bp,
                                             float* __restrict__ XW,
                                             int mtiles) {
    const int lane = threadIdx.x & 31;
    const int wid  = threadIdx.x >> 5;
    const int tile = blockIdx.x * 8 + wid;
    if (tile >= mtiles) return;                 // wave-uniform: EXEC stays all-1s

    const int m   = tile * 16 + (lane & 15);
    const int khi = (lane >> 4) * 8;
    const float* xr = X + (size_t)m * 128;

    v16h a[4];
#pragma unroll
    for (int kt = 0; kt < 4; ++kt) {
        const float* p = xr + kt * 32 + khi;    // two contiguous 8-float runs
#pragma unroll
        for (int i = 0; i < 8; ++i) a[kt][i]     = (_Float16)p[i];
#pragma unroll
        for (int i = 0; i < 8; ++i) a[kt][8 + i] = (_Float16)p[16 + i];
    }

    const int row0 = tile * 16 + (lane >> 4) * 8;
    for (int nt = 0; nt < 9; ++nt) {
        v8f acc = {};
#pragma unroll
        for (int kt = 0; kt < 4; ++kt) {
            v16h b = *reinterpret_cast<const v16h*>(Bp + ((size_t)(kt * 9 + nt) * 32 + lane) * 16);
            acc = __builtin_amdgcn_wmma_f32_16x16x32_f16(
                    false, a[kt], false, b, (short)0, acc, false, false);
        }
        const int col = nt * 16 + (lane & 15);
#pragma unroll
        for (int v = 0; v < 8; ++v)
            XW[(size_t)(row0 + v) * 144 + col] = acc[v];
    }
}

// ---------------------------------------------------------------------------
// Edge scatter-aggregate: 16 lanes per edge (one per output channel).
// Layer1: SRCF = XW1 (stride 144, per-relation offset r*16)
// Layer2: SRCF = H   (stride 16,  no relation offset)
// ---------------------------------------------------------------------------
__global__ __launch_bounds__(256) void edge_agg(const int* __restrict__ ei,
                                                const int* __restrict__ et,
                                                const float* __restrict__ SRCF,
                                                int srcStride, int perRel,
                                                float* __restrict__ AGG,
                                                float* __restrict__ CNT,
                                                int nE, int withCnt) {
    long long tid = (long long)blockIdx.x * 256 + threadIdx.x;
    int e = (int)(tid >> 4);
    if (e >= nE) return;
    int o = (int)(tid & 15);
    int s = ei[e];
    int d = ei[(size_t)nE + e];
    int r = et[e];
    float v = SRCF[(size_t)s * srcStride + (perRel ? (r << 4) : 0) + o];
    unsafeAtomicAdd(&AGG[((size_t)d * N_REL + r) * 16 + o], v);
    if (withCnt && o == 0)
        unsafeAtomicAdd(&CNT[(size_t)d * N_REL + r], 1.0f);
}

// ---------------------------------------------------------------------------
// Layer-1 epilogue: mean per (node,rel), sum over rel, + root + bias, ReLU
// ---------------------------------------------------------------------------
__global__ __launch_bounds__(256) void finish1(const float* __restrict__ AGG,
                                               const float* __restrict__ CNT,
                                               const float* __restrict__ XW,
                                               const float* __restrict__ b1,
                                               float* __restrict__ H,
                                               int nNodes) {
    long long tid = (long long)blockIdx.x * 256 + threadIdx.x;
    int n = (int)(tid >> 4);
    if (n >= nNodes) return;
    int o = (int)(tid & 15);
    float s = 0.0f;
#pragma unroll
    for (int r = 0; r < N_REL; ++r)
        s += AGG[((size_t)n * N_REL + r) * 16 + o] /
             fmaxf(CNT[(size_t)n * N_REL + r], 1.0f);
    float h = s + XW[(size_t)n * 144 + 128 + o] + b1[o];
    H[(size_t)n * 16 + o] = fmaxf(h, 0.0f);
}

// ---------------------------------------------------------------------------
// Build layer-2 A operand [100000,160] f16:
// k<128: mean-aggregated h per (rel,ch); 128<=k<144: h itself; else zero-pad
// ---------------------------------------------------------------------------
__global__ __launch_bounds__(256) void prep2(const float* __restrict__ AGG,
                                             const float* __restrict__ CNT,
                                             const float* __restrict__ H,
                                             _Float16* __restrict__ A2,
                                             int nNodes) {
    long long tid = (long long)blockIdx.x * 256 + threadIdx.x;
    int n = (int)(tid / 160);
    if (n >= nNodes) return;
    int k = (int)(tid % 160);
    float v = 0.0f;
    if (k < 128)
        v = AGG[(size_t)n * 128 + k] / fmaxf(CNT[(size_t)n * N_REL + (k >> 4)], 1.0f);
    else if (k < 144)
        v = H[(size_t)n * 16 + (k - 128)];
    A2[(size_t)n * 160 + k] = (_Float16)v;
}

// ---------------------------------------------------------------------------
// GEMM2: OUT[100000,64] = A2[100000,160] * Bcat2[160,64] + b2
// ---------------------------------------------------------------------------
__global__ __launch_bounds__(256) void gemm2(const _Float16* __restrict__ A2,
                                             const _Float16* __restrict__ Bp,
                                             const float* __restrict__ b2,
                                             float* __restrict__ OUT,
                                             int mtiles) {
    const int lane = threadIdx.x & 31;
    const int wid  = threadIdx.x >> 5;
    const int tile = blockIdx.x * 8 + wid;
    if (tile >= mtiles) return;

    const int m   = tile * 16 + (lane & 15);
    const int khi = (lane >> 4) * 8;
    const _Float16* ar = A2 + (size_t)m * 160;

    v16h a[5];
#pragma unroll
    for (int kt = 0; kt < 5; ++kt) {
        v8h h0 = *reinterpret_cast<const v8h*>(ar + kt * 32 + khi);
        v8h h1 = *reinterpret_cast<const v8h*>(ar + kt * 32 + khi + 16);
#pragma unroll
        for (int i = 0; i < 8; ++i) { a[kt][i] = h0[i]; a[kt][8 + i] = h1[i]; }
    }

    const int row0 = tile * 16 + (lane >> 4) * 8;
    for (int nt = 0; nt < 4; ++nt) {
        v8f acc = {};
#pragma unroll
        for (int kt = 0; kt < 5; ++kt) {
            v16h b = *reinterpret_cast<const v16h*>(Bp + ((size_t)(kt * 4 + nt) * 32 + lane) * 16);
            acc = __builtin_amdgcn_wmma_f32_16x16x32_f16(
                    false, a[kt], false, b, (short)0, acc, false, false);
        }
        const int col  = nt * 16 + (lane & 15);
        const float bi = b2[col];
#pragma unroll
        for (int v = 0; v < 8; ++v)
            OUT[(size_t)(row0 + v) * 64 + col] = acc[v] + bi;
    }
}

// ---------------------------------------------------------------------------
extern "C" void kernel_launch(void* const* d_in, const int* in_sizes, int n_in,
                              void* d_out, int out_size, void* d_ws, size_t ws_size,
                              hipStream_t stream) {
    const float* x     = (const float*)d_in[0];
    const int*   ei    = (const int*)d_in[1];
    const int*   et    = (const int*)d_in[2];
    const float* W1    = (const float*)d_in[3];
    const float* root1 = (const float*)d_in[4];
    const float* b1    = (const float*)d_in[5];
    const float* W2    = (const float*)d_in[6];
    const float* root2 = (const float*)d_in[7];
    const float* b2    = (const float*)d_in[8];
    float* out = (float*)d_out;

    const int nNodes = in_sizes[0] / 128;          // 100000
    const int nE     = in_sizes[1] / 2;            // 3200000
    const int mtiles = nNodes / 16;                // 6250 (exact)

    // Workspace layout (256B aligned); A2 aliases XW1 (XW1 dead after finish1)
    char* ws = (char*)d_ws;
    size_t off = 0;
    auto take = [&](size_t bytes) { char* p = ws + off; off = (off + bytes + 255) & ~(size_t)255; return p; };
    float*    AGG = (float*)   take((size_t)nNodes * 128 * 4);   // 51.2 MB
    float*    CNT = (float*)   take((size_t)nNodes * 8 * 4);     //  3.2 MB
    float*    H   = (float*)   take((size_t)nNodes * 16 * 4);    //  6.4 MB
    float*    XW1 = (float*)   take((size_t)nNodes * 144 * 4);   // 57.6 MB
    _Float16* A2  = (_Float16*)XW1;                              // alias (32 MB < 57.6 MB)
    _Float16* Bp1 = (_Float16*)take(4 * 9 * 512 * 2);
    _Float16* Bp2 = (_Float16*)take(5 * 4 * 512 * 2);

    dim3 blk(256);

    // Pack both weight operands (tiny)
    pack_b1<<<dim3((4 * 9 * 512 + 255) / 256), blk, 0, stream>>>(W1, root1, Bp1);
    pack_b2<<<dim3((5 * 4 * 512 + 255) / 256), blk, 0, stream>>>(W2, root2, Bp2);

    // Zero aggregation buffers (+ counts, computed once for both layers)
    long long nAgg = (long long)nNodes * 128;
    long long nCnt = (long long)nNodes * 8;
    zero_f32<<<dim3((unsigned)((nAgg + 255) / 256)), blk, 0, stream>>>(AGG, nAgg);
    zero_f32<<<dim3((unsigned)((nCnt + 255) / 256)), blk, 0, stream>>>(CNT, nCnt);

    // Layer 1: fused (all relations + root) GEMM via WMMA
    gemm1<<<dim3((mtiles + 7) / 8), blk, 0, stream>>>(x, Bp1, XW1, mtiles);

    // Layer 1 edge scatter (per-relation 16-wide messages) + counts
    long long eThreads = (long long)nE * 16;
    edge_agg<<<dim3((unsigned)((eThreads + 255) / 256)), blk, 0, stream>>>(
        ei, et, XW1, 144, 1, AGG, CNT, nE, 1);

    // Mean + sum over relations + root + bias + ReLU
    long long fThreads = (long long)nNodes * 16;
    finish1<<<dim3((unsigned)((fThreads + 255) / 256)), blk, 0, stream>>>(
        AGG, CNT, XW1, b1, H, nNodes);

    // Layer 2: aggregate h BEFORE transform (linearity of mean) -> 16 floats/edge
    zero_f32<<<dim3((unsigned)((nAgg + 255) / 256)), blk, 0, stream>>>(AGG, nAgg);
    edge_agg<<<dim3((unsigned)((eThreads + 255) / 256)), blk, 0, stream>>>(
        ei, et, H, 16, 0, AGG, CNT, nE, 0);

    // Build A2 = [mean-agg per (rel,ch) | h | 0-pad] in f16 (aliases XW1)
    long long pThreads = (long long)nNodes * 160;
    prep2<<<dim3((unsigned)((pThreads + 255) / 256)), blk, 0, stream>>>(
        AGG, CNT, H, A2, nNodes);

    // Layer 2 fused GEMM (sum over relations folded into K) + root + bias
    gemm2<<<dim3((mtiles + 7) / 8), blk, 0, stream>>>(A2, Bp2, b2, out, mtiles);
}
